// TrajFM_39659728011645
// MI455X (gfx1250) — matrix operand order repaired and checked by reading, not compiled, not run on hardware
//
#include <hip/hip_runtime.h>
#include <hip/hip_bf16.h>
#include <math.h>

// ---------------- model constants ----------------
#define BB    8
#define LL    256
#define BL    2048      // B*L
#define DD    128
#define NHH   8
#define DHH   16
#define FFD   256
#define NPOI  20000
#define CHUNK 2048      // POI coords per LDS tile (16 KB)
#define PADTK 5

typedef __attribute__((ext_vector_type(16))) _Float16 v16h;
typedef __attribute__((ext_vector_type(8)))  float    v8f;

__device__ __forceinline__ float leakyf(float x) { return x >= 0.f ? x : 0.01f * x; }

// Block-wide LayerNorm normalization for a 128-wide row (1 elem / thread).
__device__ __forceinline__ float ln_norm128(float x, float* red) {
  const int t = threadIdx.x;
  red[t] = x; __syncthreads();
  for (int s = 64; s > 0; s >>= 1) { if (t < s) red[t] += red[t + s]; __syncthreads(); }
  float mean = red[0] * (1.f / 128.f); __syncthreads();
  float d = x - mean;
  red[t] = d * d; __syncthreads();
  for (int s = 64; s > 0; s >>= 1) { if (t < s) red[t] += red[t + s]; __syncthreads(); }
  float var = red[0] * (1.f / 128.f); __syncthreads();
  return d * rsqrtf(var + 1e-5f);
}

// ---------------------------------------------------------------------------
// Weight pre-pack: f32 K x N row-major  ->  f16 tiles in WMMA B-operand order.
// Tile (kt, nt): 32 lanes x 16 halves; lane's halves i=0..15 hold
//   W[(kt*32 + (lane>>4)*16 + i) * N + nt*16 + (lane&15)]
// grid = (N/16, K/32), block = 32.  One 32B v16h store per lane.
// ---------------------------------------------------------------------------
__global__ __launch_bounds__(32) void pack_w_kernel(
    const float* __restrict__ W, _Float16* __restrict__ Wp, int N)
{
  const int nt = blockIdx.x, kt = blockIdx.y, ntiles = gridDim.x;
  const int lane = threadIdx.x & 31;
  const int mr = lane & 15, halfId = lane >> 4;
  v16h b;
#pragma unroll
  for (int i = 0; i < 16; ++i)
    b[i] = (_Float16)W[(size_t)(kt * 32 + halfId * 16 + i) * N + nt * 16 + mr];
  *(v16h*)(Wp + ((size_t)(kt * ntiles + nt) * 32 + lane) * 16) = b;
}

// ---------------------------------------------------------------------------
// WMMA GEMM:  C[M x N] = act( A[M x K] @ W[K x N] + bias ) (+ resid)
// K templated & fully unrolled -> K/32 back-to-back v_wmma per wave.
// A: f32, loaded as 4x float4 (b128) per k-step per lane, cvt to f16.
// W: pre-packed f16 tiles, one 32B v16h load per k-step per lane.
// grid = (N/16, M/16), block = 32 (one wave per 16x16 C tile).
// ---------------------------------------------------------------------------
template <int K, bool RESID, int ACT>
__global__ __launch_bounds__(32) void wmma_gemm_kernel(
    const float* __restrict__ A, int lda,
    const _Float16* __restrict__ Wp,
    const float* __restrict__ bias,
    const float* __restrict__ resid, int ldr,
    float* __restrict__ C, int ldc)
{
  const int lane   = threadIdx.x & 31;
  const int nt     = blockIdx.x;
  const int ntiles = gridDim.x;
  const int m0     = blockIdx.y * 16;
  const int n0     = nt * 16;
  const int mr     = lane & 15;
  const int halfId = lane >> 4;

  const float* Arow = A + (size_t)(m0 + mr) * lda;

  v8f acc = {};
#pragma unroll
  for (int kt = 0; kt < K / 32; ++kt) {
    // A fragment: halves 0..15 <- A[kt*32 + halfId*16 + 0..15]
    const float4 c0 = *(const float4*)(Arow + kt * 32 + halfId * 16);
    const float4 c1 = *(const float4*)(Arow + kt * 32 + halfId * 16 + 4);
    const float4 c2 = *(const float4*)(Arow + kt * 32 + halfId * 16 + 8);
    const float4 c3 = *(const float4*)(Arow + kt * 32 + halfId * 16 + 12);
    v16h a;
    a[0]  = (_Float16)c0.x; a[1]  = (_Float16)c0.y; a[2]  = (_Float16)c0.z; a[3]  = (_Float16)c0.w;
    a[4]  = (_Float16)c1.x; a[5]  = (_Float16)c1.y; a[6]  = (_Float16)c1.z; a[7]  = (_Float16)c1.w;
    a[8]  = (_Float16)c2.x; a[9]  = (_Float16)c2.y; a[10] = (_Float16)c2.z; a[11] = (_Float16)c2.w;
    a[12] = (_Float16)c3.x; a[13] = (_Float16)c3.y; a[14] = (_Float16)c3.z; a[15] = (_Float16)c3.w;

    const v16h b = *(const v16h*)(Wp + ((size_t)(kt * ntiles + nt) * 32 + lane) * 16);

    acc = __builtin_amdgcn_wmma_f32_16x16x32_f16(false, a, false, b,
                                                 (short)0, acc, false, false);
  }

  const int col = n0 + mr;
  const float bv = bias[col];
#pragma unroll
  for (int r = 0; r < 8; ++r) {
    int row = m0 + halfId * 8 + r;     // C/D layout: VGPR r holds rows M and M+8
    float val = acc[r] + bv;
    if constexpr (RESID) val += resid[(size_t)row * ldr + col];
    if constexpr (ACT == 1) val = fmaxf(val, 0.f);
    if constexpr (ACT == 2) val = leakyf(val);
    C[(size_t)row * ldc + col] = val;
  }
}

// ---------------------------------------------------------------------------
// Precompute te_table[6][128] = LN(tok_emb) @ tok_w + tok_b
// ---------------------------------------------------------------------------
__global__ __launch_bounds__(128) void te_table_kernel(
    const float* __restrict__ tok_emb, const float* __restrict__ g,
    const float* __restrict__ bln, const float* __restrict__ tok_w,
    const float* __restrict__ tok_b, float* __restrict__ te_table)
{
  __shared__ float red[128];
  __shared__ float lnv[128];
  const int t = threadIdx.x;
  for (int tok = 0; tok < 6; ++tok) {
    float x = tok_emb[tok * 128 + t];
    float nv = ln_norm128(x, red) * g[t] + bln[t];
    lnv[t] = nv; __syncthreads();
    float dot = 0.f;
#pragma unroll 4
    for (int e = 0; e < 128; ++e) dot += lnv[e] * tok_w[e * 128 + t];
    te_table[tok * 128 + t] = dot + tok_b[t];
    __syncthreads();
  }
}

// ---------------------------------------------------------------------------
// Per-point prep: tokens, spatial hidden (leaky(sp_w1)), fourier features.
// grid = BL, block = 128.
// ---------------------------------------------------------------------------
__global__ __launch_bounds__(128) void prep_kernel(
    const float* __restrict__ iseq,
    const float* __restrict__ sp_w1, const float* __restrict__ sp_b1,
    const float* __restrict__ f_omega, const float* __restrict__ f_phi,
    float* __restrict__ h_sp, float* __restrict__ fe,
    int* __restrict__ tok0, int* __restrict__ tok1)
{
  const int p = blockIdx.x, n = threadIdx.x;
  const float* ip = iseq + (size_t)p * 8;   // [c][j] -> 2c + j
  float sx = ip[0], sy = ip[2], ts = ip[4], dts = ip[6];
  if (n == 0) { tok0[p] = (int)ip[1]; tok1[p] = (int)ip[5]; }

  float h = sx * sp_w1[n] + sy * sp_w1[128 + n] + sp_b1[n];
  h_sp[(size_t)p * 128 + n] = leakyf(h);

  float tt0 = fmodf(ts, 604800.f) * (1.f / 86400.f);
  float tt1 = fmodf(ts, 86400.f)  * (1.f / 3600.f);
  float tt2 = fmodf(ts, 3600.f)   * (1.f / 60.f);
  float tt3 = dts * (1.f / 60.f);
  float tt[4] = {tt0, tt1, tt2, tt3};
#pragma unroll
  for (int j = 0; j < 4; ++j) {
    float v = cosf(tt[j] * f_omega[j * 128 + n] + f_phi[j * 128 + n]);
    fe[(size_t)p * 512 + j * 128 + n] = leakyf(v);
  }
}

// ---------------------------------------------------------------------------
// POI nearest neighbor: 2048 points x 20000 POIs; POI coords tiled in LDS.
// grid = 8, block = 256 (one thread = one point).
// ---------------------------------------------------------------------------
__global__ __launch_bounds__(256) void poi_argmin_kernel(
    const float* __restrict__ iseq, const float* __restrict__ poi,
    int* __restrict__ poi_idx)
{
  __shared__ float sx[CHUNK];
  __shared__ float sy[CHUNK];
  const int t = threadIdx.x;
  const int p = blockIdx.x * 256 + t;
  const float px = iseq[(size_t)p * 8 + 0];
  const float py = iseq[(size_t)p * 8 + 2];

  float best = 3.4e38f; int bi = 0;
  for (int c0 = 0; c0 < NPOI; c0 += CHUNK) {
    for (int i = t; i < CHUNK; i += 256) {
      int g = c0 + i;
      if (g < NPOI) { sx[i] = poi[g * 2]; sy[i] = poi[g * 2 + 1]; }
      else          { sx[i] = 1e18f;      sy[i] = 0.f; }
    }
    if (c0 + CHUNK < NPOI)
      __builtin_prefetch(&poi[(size_t)(c0 + CHUNK) * 2 + t * 2], 0, 1);
    __syncthreads();
#pragma unroll 4
    for (int i = 0; i < CHUNK; ++i) {
      float x = sx[i], y = sy[i];
      float d = x * x + y * y - 2.f * (px * x + py * y);  // matches reference d2
      if (d < best) { best = d; bi = c0 + i; }            // first-min tie break
    }
    __syncthreads();
  }
  poi_idx[p] = bi;
}

// LN of gathered poi_embed rows.  grid = BL, block = 128.
__global__ __launch_bounds__(128) void ln_gather_kernel(
    const float* __restrict__ table, const int* __restrict__ idx,
    const float* __restrict__ g, const float* __restrict__ b,
    float* __restrict__ Y)
{
  __shared__ float red[128];
  const int t = threadIdx.x;
  const int row = idx[blockIdx.x];
  float x = table[(size_t)row * 128 + t];
  Y[(size_t)blockIdx.x * 128 + t] = ln_norm128(x, red) * g[t] + b[t];
}

// Row LayerNorm.  grid = rows, block = 128.
__global__ __launch_bounds__(128) void ln_kernel(
    const float* __restrict__ X, const float* __restrict__ g,
    const float* __restrict__ b, float* __restrict__ Y)
{
  __shared__ float red[128];
  const int t = threadIdx.x;
  const size_t row = blockIdx.x;
  float x = X[row * 128 + t];
  Y[row * 128 + t] = ln_norm128(x, red) * g[t] + b[t];
}

// Add token-embedding table rows into the 3 modal slots.  grid = BL, block = 128.
__global__ __launch_bounds__(128) void finalize_modal_kernel(
    float* __restrict__ modal, const float* __restrict__ te_table,
    const int* __restrict__ tok0, const int* __restrict__ tok1)
{
  const int p = blockIdx.x, n = threadIdx.x;
  const size_t base = (size_t)p * 3 * 128;
  float t0 = te_table[tok0[p] * 128 + n];
  float t1 = te_table[tok1[p] * 128 + n];
  modal[base + 0 * 128 + n] += t0;   // spatial_e  + te[:,:,0]
  modal[base + 1 * 128 + n] += t1;   // temporal_e + te[:,:,1]
  modal[base + 2 * 128 + n] += t0;   // poi_e      + te[:,:,0]
}

// Mixer attention: per point, 8 heads over 3 tokens.  grid = BL, block = 128.
__global__ __launch_bounds__(128) void mixer_attn_kernel(
    const float* __restrict__ Q, const float* __restrict__ K,
    const float* __restrict__ V, float* __restrict__ O)
{
  __shared__ float sS[8][9];
  const int p = blockIdx.x, t = threadIdx.x;
  const size_t base = (size_t)p * 3 * 128;
  if (t < 72) {
    int h = t / 9, ij = t % 9, i = ij / 3, j = ij % 3;
    float dot = 0.f;
#pragma unroll
    for (int d = 0; d < 16; ++d)
      dot += Q[base + i * 128 + h * 16 + d] * K[base + j * 128 + h * 16 + d];
    sS[h][i * 3 + j] = dot * 0.25f;   // 1/sqrt(16)
  }
  __syncthreads();
  if (t < 24) {
    int h = t / 3, i = t % 3;
    float s0 = sS[h][i * 3 + 0], s1 = sS[h][i * 3 + 1], s2 = sS[h][i * 3 + 2];
    float m = fmaxf(s0, fmaxf(s1, s2));
    float e0 = __expf(s0 - m), e1 = __expf(s1 - m), e2 = __expf(s2 - m);
    float inv = 1.f / (e0 + e1 + e2);
    sS[h][i * 3 + 0] = e0 * inv; sS[h][i * 3 + 1] = e1 * inv; sS[h][i * 3 + 2] = e2 * inv;
  }
  __syncthreads();
  const int h = t / 16;
#pragma unroll
  for (int i = 0; i < 3; ++i) {
    float o = 0.f;
#pragma unroll
    for (int j = 0; j < 3; ++j)
      o += sS[h][i * 3 + j] * V[base + j * 128 + t];
    O[base + i * 128 + t] = o;
  }
}

// Mean over modal tokens + sinusoidal position encodings.  grid = BL, block = 128.
__global__ __launch_bounds__(128) void meanpos_kernel(
    const float* __restrict__ mixo, const int* __restrict__ positions,
    float* __restrict__ modal_h, float* __restrict__ out0)
{
  const int p = blockIdx.x, n = threadIdx.x;
  const size_t base = (size_t)p * 3 * 128;
  float v = (mixo[base + n] + mixo[base + 128 + n] + mixo[base + 256 + n]) * (1.f / 3.f);
  const int i = n >> 1;
  float div = powf(10000.f, (float)(2 * i) * (1.f / 128.f));
  float a0 = (float)positions[p * 2 + 0] / div;
  float a1 = (float)positions[p * 2 + 1] / div;
  v += ((n & 1) ? cosf(a0) : sinf(a0)) + ((n & 1) ? cosf(a1) : sinf(a1));
  modal_h[(size_t)p * 128 + n] = v;
  out0[(size_t)p * 128 + n]    = v;
}

// RoPE on Q and K (dh=16, 8 heads).  grid = BL, block = 64.
__global__ __launch_bounds__(64) void rope_kernel(float* __restrict__ Q,
                                                  float* __restrict__ K)
{
  const int row = blockIdx.x;
  const int l = row % LL;
  const int t = threadIdx.x, h = t / 8, i = t % 8;
  float inv = powf(10000.f, -(float)(2 * i) * (1.f / 16.f));
  float ang = (float)l * inv;
  float c = cosf(ang), s = sinf(ang);
  size_t idx = (size_t)row * 128 + h * 16 + 2 * i;
  float q1 = Q[idx], q2 = Q[idx + 1];
  Q[idx] = q1 * c - q2 * s; Q[idx + 1] = q1 * s + q2 * c;
  float k1 = K[idx], k2 = K[idx + 1];
  K[idx] = k1 * c - k2 * s; K[idx + 1] = k1 * s + k2 * c;
}

// Sequence attention, one (b,h) per block, K/V in LDS, online softmax.
// grid = B*NH, block = 256 (thread = query position l).
__global__ __launch_bounds__(256) void seq_attn_kernel(
    const float* __restrict__ Q, const float* __restrict__ K,
    const float* __restrict__ V, const int* __restrict__ tok0,
    float* __restrict__ O)
{
  __shared__ float sK[256 * 16];
  __shared__ float sV[256 * 16];
  __shared__ int   sPad[256];
  const int bh = blockIdx.x;
  const int b = bh / NHH, h = bh % NHH;
  const int t = threadIdx.x;
  const size_t rb = (size_t)(b * LL + t) * 128 + h * 16;
#pragma unroll
  for (int d = 0; d < 16; ++d) { sK[t * 16 + d] = K[rb + d]; sV[t * 16 + d] = V[rb + d]; }
  sPad[t] = (tok0[b * LL + t] == PADTK) ? 1 : 0;
  __syncthreads();

  float q[16];
#pragma unroll
  for (int d = 0; d < 16; ++d) q[d] = Q[rb + d];

  float m = -1e30f, sum = 0.f;
  float acc[16];
#pragma unroll
  for (int d = 0; d < 16; ++d) acc[d] = 0.f;

  for (int j = 0; j < 256; ++j) {
    float sc;
    if (j > t || sPad[j]) sc = -1e9f;                 // same constant as reference
    else {
      float dot = 0.f;
#pragma unroll
      for (int d = 0; d < 16; ++d) dot += q[d] * sK[j * 16 + d];
      sc = dot * 0.25f;
    }
    float nm = fmaxf(m, sc);
    float corr = __expf(m - nm);
    float e = __expf(sc - nm);
    sum = sum * corr + e;
#pragma unroll
    for (int d = 0; d < 16; ++d) acc[d] = acc[d] * corr + e * sV[j * 16 + d];
    m = nm;
  }
  float inv = 1.f / sum;
#pragma unroll
  for (int d = 0; d < 16; ++d) O[rb + d] = acc[d] * inv;
}

__global__ void copy_kernel(const float* __restrict__ s, float* __restrict__ d, int n)
{
  int i = blockIdx.x * blockDim.x + threadIdx.x;
  if (i < n) d[i] = s[i];
}

// ---------------------------------------------------------------------------
// Host side
// ---------------------------------------------------------------------------
struct LayerP {
  const float *b1,*b2,*bk,*bo,*bq,*bv,*ln1_b,*ln1_g,*ln2_b,*ln2_g,*w1,*w2,*wk,*wo,*wq,*wv;
};
struct LayerW {            // packed f16 weights
  _Float16 *wq,*wk,*wv,*wo,*w1,*w2;
};

template <int K, bool RESID, int ACT>
static inline void launch_gemm_t(hipStream_t s, const float* A, int lda,
                                 const _Float16* Wp, const float* bias,
                                 const float* resid, int ldr,
                                 float* C, int ldc, int M, int N)
{
  dim3 grid(N / 16, M / 16);
  wmma_gemm_kernel<K, RESID, ACT><<<grid, 32, 0, s>>>(A, lda, Wp, bias, resid, ldr, C, ldc);
}

extern "C" void kernel_launch(void* const* d_in, const int* in_sizes, int n_in,
                              void* d_out, int out_size, void* d_ws, size_t ws_size,
                              hipStream_t stream)
{
  auto F = [&](int i) -> const float* { return (const float*)d_in[i]; };

  // -------- input pointer resolution (handles both flatten orders) --------
  // Insertion order: input_seq, positions, poi_coors, <params sorted leaves>
  // Sorted order:    input_seq, <params sorted leaves>, poi_coors, positions
  const float* iseq = F(0);
  const int* positions; const float* poi_coors; int pb;
  if (in_sizes[1] == 2 * BB * LL) {                 // positions at slot 1
    positions = (const int*)d_in[1]; poi_coors = F(2); pb = 3;
  } else {
    pb = 1; poi_coors = F(67); positions = (const int*)d_in[68];
  }
  const float* f_omega  = F(pb + 0);
  const float* f_phi    = F(pb + 1);
  const int    mb       = pb + 2;                   // mixer: 16 sorted leaves
  const float* poi_b    = F(pb + 18);
  const float* poi_emb  = F(pb + 19);
  const float* poi_ln_b = F(pb + 20);
  const float* poi_ln_g = F(pb + 21);
  const float* poi_w    = F(pb + 22);
  const int    s0b      = pb + 23;                  // seq[0]
  const int    s1b      = pb + 39;                  // seq[1]
  const float* sp_b1    = F(pb + 55);
  const float* sp_b2    = F(pb + 56);
  const float* sp_w1    = F(pb + 57);
  const float* sp_w2    = F(pb + 58);
  const float* t_b      = F(pb + 59);
  const float* t_w      = F(pb + 60);
  const float* tok_b    = F(pb + 61);
  const float* tok_emb  = F(pb + 62);
  const float* tok_ln_b = F(pb + 63);
  const float* tok_ln_g = F(pb + 64);
  const float* tok_w    = F(pb + 65);

  auto mkLayer = [&](int base) {
    LayerP P;
    P.b1 = F(base + 0);  P.b2 = F(base + 1);  P.bk = F(base + 2);  P.bo = F(base + 3);
    P.bq = F(base + 4);  P.bv = F(base + 5);  P.ln1_b = F(base + 6); P.ln1_g = F(base + 7);
    P.ln2_b = F(base + 8); P.ln2_g = F(base + 9); P.w1 = F(base + 10); P.w2 = F(base + 11);
    P.wk = F(base + 12); P.wo = F(base + 13); P.wq = F(base + 14); P.wv = F(base + 15);
    return P;
  };
  LayerP mixer = mkLayer(mb), seq0 = mkLayer(s0b), seq1 = mkLayer(s1b);

  // -------- workspace carve --------
  int* wsI = (int*)d_ws;
  int* TOK0 = wsI;            // 2048
  int* TOK1 = wsI + 2048;     // 2048
  int* PIDX = wsI + 4096;     // 2048
  float* wsF = (float*)(wsI + 6144);
  size_t o = 0;
  float* TE_TAB  = wsF + o; o += 6 * 128 + 64;              // keep 16B-multiples
  float* H_SP    = wsF + o; o += (size_t)BL * 128;
  float* FE      = wsF + o; o += (size_t)BL * 512;
  float* POI_LN  = wsF + o; o += (size_t)BL * 128;
  float* MODAL   = wsF + o; o += (size_t)3 * BL * 128;
  float* BUF_Q   = wsF + o; o += (size_t)3 * BL * 128;
  float* BUF_K   = wsF + o; o += (size_t)3 * BL * 128;
  float* BUF_V   = wsF + o; o += (size_t)3 * BL * 128;
  float* BUF_O   = wsF + o; o += (size_t)3 * BL * 128;
  float* BUF_T   = wsF + o; o += (size_t)3 * BL * 128;
  float* BUF_H   = wsF + o; o += (size_t)3 * BL * 256;
  float* XCUR    = wsF + o; o += (size_t)BL * 128;          // modal_h / running x

  // packed f16 weights region (after the float region; 32B aligned by construction)
  _Float16* wsH = (_Float16*)(wsF + o);
  size_t oh = 0;
  auto allocH = [&](size_t n) { _Float16* p = wsH + oh; oh += n; return p; };
  _Float16* PW_SP2  = allocH(128 * 128);
  _Float16* PW_TW   = allocH(512 * 128);
  _Float16* PW_POIW = allocH(128 * 128);
  auto mkLayerW = [&]() {
    LayerW W;
    W.wq = allocH(128 * 128); W.wk = allocH(128 * 128); W.wv = allocH(128 * 128);
    W.wo = allocH(128 * 128); W.w1 = allocH(128 * 256); W.w2 = allocH(256 * 128);
    return W;
  };
  LayerW mixW = mkLayerW(), s0W = mkLayerW(), s1W = mkLayerW();

  float* out0 = (float*)d_out;                  // modal_h (B,L,D)
  float* out1 = out0 + (size_t)BL * 128;        // mem     (B,L,D)

  // -------- stage 0: pack all weights to f16 WMMA tile order --------
  auto pack = [&](const float* W, _Float16* Wp, int K, int N) {
    dim3 g(N / 16, K / 32);
    pack_w_kernel<<<g, 32, 0, stream>>>(W, Wp, N);
  };
  pack(sp_w2, PW_SP2, 128, 128);
  pack(t_w,   PW_TW,  512, 128);
  pack(poi_w, PW_POIW,128, 128);
  const LayerP* lp[3] = { &mixer, &seq0, &seq1 };
  LayerW*       lw[3] = { &mixW,  &s0W,  &s1W  };
  for (int i = 0; i < 3; ++i) {
    pack(lp[i]->wq, lw[i]->wq, 128, 128); pack(lp[i]->wk, lw[i]->wk, 128, 128);
    pack(lp[i]->wv, lw[i]->wv, 128, 128); pack(lp[i]->wo, lw[i]->wo, 128, 128);
    pack(lp[i]->w1, lw[i]->w1, 128, 256); pack(lp[i]->w2, lw[i]->w2, 256, 128);
  }

  // -------- stage 1: embeddings --------
  te_table_kernel<<<1, 128, 0, stream>>>(tok_emb, tok_ln_g, tok_ln_b, tok_w, tok_b, TE_TAB);
  prep_kernel<<<BL, 128, 0, stream>>>(iseq, sp_w1, sp_b1, f_omega, f_phi, H_SP, FE, TOK0, TOK1);
  poi_argmin_kernel<<<BL / 256, 256, 0, stream>>>(iseq, poi_coors, PIDX);
  ln_gather_kernel<<<BL, 128, 0, stream>>>(poi_emb, PIDX, poi_ln_g, poi_ln_b, POI_LN);

  // modal slots: [0]=spatial_e, [1]=temporal_e, [2]=poi_e  (row stride 384)
  launch_gemm_t<128, false, 0>(stream, H_SP,  128, PW_SP2,  sp_b2, nullptr, 0, MODAL + 0,   384, BL, 128);
  launch_gemm_t<512, false, 0>(stream, FE,    512, PW_TW,   t_b,   nullptr, 0, MODAL + 128, 384, BL, 128);
  launch_gemm_t<128, false, 0>(stream, POI_LN,128, PW_POIW, poi_b, nullptr, 0, MODAL + 256, 384, BL, 128);
  finalize_modal_kernel<<<BL, 128, 0, stream>>>(MODAL, TE_TAB, TOK0, TOK1);

  // -------- stage 2: mixer encoder layer (seq len 3, M = 6144 rows) --------
  {
    const int M = 3 * BL;
    launch_gemm_t<128, false, 0>(stream, MODAL, 128, mixW.wq, mixer.bq, nullptr, 0, BUF_Q, 128, M, 128);
    launch_gemm_t<128, false, 0>(stream, MODAL, 128, mixW.wk, mixer.bk, nullptr, 0, BUF_K, 128, M, 128);
    launch_gemm_t<128, false, 0>(stream, MODAL, 128, mixW.wv, mixer.bv, nullptr, 0, BUF_V, 128, M, 128);
    mixer_attn_kernel<<<BL, 128, 0, stream>>>(BUF_Q, BUF_K, BUF_V, BUF_O);
    launch_gemm_t<128, true, 0>(stream, BUF_O, 128, mixW.wo, mixer.bo, MODAL, 128, BUF_T, 128, M, 128);
    ln_kernel<<<M, 128, 0, stream>>>(BUF_T, mixer.ln1_g, mixer.ln1_b, MODAL);
    launch_gemm_t<128, false, 1>(stream, MODAL, 128, mixW.w1, mixer.b1, nullptr, 0, BUF_H, 256, M, 256);
    launch_gemm_t<256, true, 0>(stream, BUF_H, 256, mixW.w2, mixer.b2, MODAL, 128, BUF_T, 128, M, 128);
    ln_kernel<<<M, 128, 0, stream>>>(BUF_T, mixer.ln2_g, mixer.ln2_b, MODAL);
  }

  // -------- stage 3: modal mean + positional encodings --------
  meanpos_kernel<<<BL, 128, 0, stream>>>(MODAL, positions, XCUR, out0);

  // -------- stage 4: 2 sequence layers (RoPE + causal/pad mask) --------
  const LayerP* layers[2]  = { &seq0, &seq1 };
  LayerW*       layersW[2] = { &s0W,  &s1W  };
  for (int li = 0; li < 2; ++li) {
    const LayerP& P = *layers[li];
    const LayerW& W = *layersW[li];
    launch_gemm_t<128, false, 0>(stream, XCUR, 128, W.wq, P.bq, nullptr, 0, BUF_Q, 128, BL, 128);
    launch_gemm_t<128, false, 0>(stream, XCUR, 128, W.wk, P.bk, nullptr, 0, BUF_K, 128, BL, 128);
    launch_gemm_t<128, false, 0>(stream, XCUR, 128, W.wv, P.bv, nullptr, 0, BUF_V, 128, BL, 128);
    rope_kernel<<<BL, 64, 0, stream>>>(BUF_Q, BUF_K);
    seq_attn_kernel<<<BB * NHH, 256, 0, stream>>>(BUF_Q, BUF_K, BUF_V, TOK0, BUF_O);
    launch_gemm_t<128, true, 0>(stream, BUF_O, 128, W.wo, P.bo, XCUR, 128, BUF_T, 128, BL, 128);
    ln_kernel<<<BL, 128, 0, stream>>>(BUF_T, P.ln1_g, P.ln1_b, XCUR);
    launch_gemm_t<128, false, 1>(stream, XCUR, 128, W.w1, P.b1, nullptr, 0, BUF_H, 256, BL, 256);
    launch_gemm_t<256, true, 0>(stream, BUF_H, 256, W.w2, P.b2, XCUR, 128, BUF_T, 128, BL, 128);
    ln_kernel<<<BL, 128, 0, stream>>>(BUF_T, P.ln2_g, P.ln2_b, XCUR);
  }

  copy_kernel<<<(BL * 128 + 255) / 256, 256, 0, stream>>>(XCUR, out1, BL * 128);
}
